// SelfAttention_1030792151600
// MI455X (gfx1250) — matrix operand order repaired and checked by reading, not compiled, not run on hardware
//
#include <hip/hip_runtime.h>
#include <math.h>

// Problem dims (fixed by the reference setup)
#define BB 16
#define CC 256
#define NN 4096   // W*H = 64*64
#define CQ 32     // C/8

typedef __attribute__((ext_vector_type(16))) __bf16 v16bf;
typedef __attribute__((ext_vector_type(8)))  __bf16 v8bf;
typedef __attribute__((ext_vector_type(8)))  float  v8f;

// Build a 16-element bf16 operand from two contiguous 16B chunks.
// (Matches CDNA5 WMMA A/B VGPR striping: each lane supplies two 16B pieces
//  whose position depends on the lane half.)
static __device__ __forceinline__ v16bf ld2(const __bf16* p0, const __bf16* p1) {
  v16bf r;
  *reinterpret_cast<v8bf*>(&r)       = *reinterpret_cast<const v8bf*>(p0);
  *(reinterpret_cast<v8bf*>(&r) + 1) = *reinterpret_cast<const v8bf*>(p1);
  return r;
}

static __device__ __forceinline__ v8f wmma_bf16(v16bf a, v16bf b, v8f c) {
  // (neg_a, A, neg_b, B, c_mod, C, reuse_a, reuse_b)
  return __builtin_amdgcn_wmma_f32_16x16x32_bf16(false, a, false, b, (short)0, c,
                                                 false, false);
}

// ---------------------------------------------------------------------------
// Kernel 1: xt[b,n,c] = bf16(x[b,c,n])   (tiled LDS transpose + convert)
// ---------------------------------------------------------------------------
__global__ __launch_bounds__(256) void transpose_convert_kernel(
    const float* __restrict__ x, __bf16* __restrict__ xt) {
  __shared__ float t[32][33];
  const int b  = blockIdx.z;
  const int n0 = blockIdx.x * 32;
  const int c0 = blockIdx.y * 32;
  const int tx = threadIdx.x, ty = threadIdx.y;
#pragma unroll
  for (int j = 0; j < 4; ++j) {
    const int c = c0 + ty + j * 8;
    t[ty + j * 8][tx] = x[((size_t)b * CC + c) * NN + n0 + tx];
  }
  __syncthreads();
#pragma unroll
  for (int j = 0; j < 4; ++j) {
    const int n = n0 + ty + j * 8;
    xt[((size_t)b * NN + n) * CC + c0 + tx] = (__bf16)t[tx][ty + j * 8];
  }
}

// ---------------------------------------------------------------------------
// Kernel 2: generic f32 -> bf16 convert (weights)
// ---------------------------------------------------------------------------
__global__ void f32_to_bf16_kernel(const float* __restrict__ in,
                                   __bf16* __restrict__ out, int n) {
  const int i = blockIdx.x * blockDim.x + threadIdx.x;
  if (i < n) out[i] = (__bf16)in[i];
}

// ---------------------------------------------------------------------------
// Kernel 3: q/k projection.  outT[b,n,o] = sum_c w[o,c]*xt[b,n,c] + bias[o]
// One wave per 16x16 output tile; K = C = 256 -> 8 WMMAs.
// ---------------------------------------------------------------------------
__global__ __launch_bounds__(32) void proj_qk_kernel(
    const __bf16* __restrict__ xt, const __bf16* __restrict__ w,
    const float* __restrict__ bias, __bf16* __restrict__ outT) {
  const int lane = threadIdx.x & 31;
  const int col  = lane & 15;
  const bool hi  = lane >= 16;
  const int n0 = blockIdx.x * 16;
  const int o0 = blockIdx.y * 16;
  const int b  = blockIdx.z;

  v8f acc = {0.f, 0.f, 0.f, 0.f, 0.f, 0.f, 0.f, 0.f};
  const __bf16* xrow = xt + ((size_t)b * NN + n0 + col) * CC;  // A row n
  const __bf16* wrow = w + (size_t)(o0 + col) * CC;            // B col o
#pragma unroll
  for (int kc = 0; kc < CC; kc += 32) {
    const __bf16* ap = xrow + kc + (hi ? 8 : 0);
    v16bf A = ld2(ap, ap + 16);
    const __bf16* bp = wrow + kc + (hi ? 8 : 0);
    v16bf Bm = ld2(bp, bp + 16);
    acc = wmma_bf16(A, Bm, acc);
  }
  const float bo = bias[o0 + col];
#pragma unroll
  for (int r = 0; r < 8; ++r) {
    const int n = n0 + r + (hi ? 8 : 0);
    outT[((size_t)b * NN + n) * CQ + o0 + col] = (__bf16)(acc[r] + bo);
  }
}

// ---------------------------------------------------------------------------
// Kernel 4: v projection.  vb[b,c,n] = sum_ci wv[c,ci]*xt[b,n,ci] + bv[c]
// ---------------------------------------------------------------------------
__global__ __launch_bounds__(32) void proj_v_kernel(
    const __bf16* __restrict__ xt, const __bf16* __restrict__ wv,
    const float* __restrict__ bv, __bf16* __restrict__ vb) {
  const int lane = threadIdx.x & 31;
  const int col  = lane & 15;
  const bool hi  = lane >= 16;
  const int n0 = blockIdx.x * 16;
  const int c0 = blockIdx.y * 16;
  const int b  = blockIdx.z;

  v8f acc = {0.f, 0.f, 0.f, 0.f, 0.f, 0.f, 0.f, 0.f};
  const __bf16* wrow = wv + (size_t)(c0 + col) * CC;            // A row c_out
  const __bf16* xrow = xt + ((size_t)b * NN + n0 + col) * CC;   // B col n
#pragma unroll
  for (int kc = 0; kc < CC; kc += 32) {
    const __bf16* ap = wrow + kc + (hi ? 8 : 0);
    v16bf A = ld2(ap, ap + 16);
    const __bf16* bp = xrow + kc + (hi ? 8 : 0);
    v16bf Bm = ld2(bp, bp + 16);
    acc = wmma_bf16(A, Bm, acc);
  }
#pragma unroll
  for (int r = 0; r < 8; ++r) {
    const int c = c0 + r + (hi ? 8 : 0);
    vb[((size_t)b * CC + c) * NN + n0 + col] = (__bf16)(acc[r] + bv[c]);
  }
}

// ---------------------------------------------------------------------------
// Kernel 5: flash attention + residual.
//   energy = qT * k    (K = Cq = 32 -> one WMMA per 16x16 tile)
//   online softmax (running max / running sum, per-row, computed ONCE)
//   outT(n,c) += P(n,m) * v^T(m,c)  via WMMA, P staged through LDS
//   out[b,c,n] = gamma * outT(n,c)/l(n) + x[b,c,n]
// Block = 128 threads = 4 waves. Each wave owns a private 16-row n-tile and
// the FULL 256 channels (16 f32 accumulators = 128 VGPRs, fine under wave32's
// 1024-VGPR budget). No inter-wave communication -> no barriers; LDS P tiles
// are wave-private, fenced with s_wait_dscnt.
// Per m-chunk per wave: 2 energy WMMA + 16 PV WMMA vs one softmax pass.
// ---------------------------------------------------------------------------
__global__ __launch_bounds__(128, 1) void attn_kernel(
    const __bf16* __restrict__ qT, const __bf16* __restrict__ kT,
    const __bf16* __restrict__ vb, const float* __restrict__ x,
    const float* __restrict__ gamma, float* __restrict__ out) {
  __shared__ __attribute__((aligned(16))) __bf16 Plds[4][16][32];  // 1KB/wave

  const int lane = threadIdx.x & 31;
  const int wv   = threadIdx.x >> 5;
  const int col  = lane & 15;
  const bool hi  = lane >= 16;
  const int b  = blockIdx.y;
  const int n0 = (blockIdx.x * 4 + wv) * 16;

  // A (q) operand: row n = n0+col, K = Cq, reused for all m-chunks.
  const __bf16* qrow = qT + ((size_t)b * NN + n0 + col) * CQ + (hi ? 8 : 0);
  const v16bf Aq = ld2(qrow, qrow + 16);

  v8f acc[16] = {};
  float mrun[8], lrun[8];
#pragma unroll
  for (int r = 0; r < 8; ++r) { mrun[r] = -1e30f; lrun[r] = 0.f; }

  const v8f zero = {0.f, 0.f, 0.f, 0.f, 0.f, 0.f, 0.f, 0.f};

  for (int m0 = 0; m0 < NN; m0 += 32) {
    // B (k) operands: columns m0..m0+15 and m0+16..m0+31
    const __bf16* k0p = kT + ((size_t)b * NN + m0 + col) * CQ + (hi ? 8 : 0);
    const __bf16* k1p = kT + ((size_t)b * NN + m0 + 16 + col) * CQ + (hi ? 8 : 0);
    v16bf Bk0 = ld2(k0p, k0p + 16);
    v16bf Bk1 = ld2(k1p, k1p + 16);
    v8f E0 = wmma_bf16(Aq, Bk0, zero);
    v8f E1 = wmma_bf16(Aq, Bk1, zero);

    // Online softmax: per-row (VGPR r x lane-half) stats; 16-lane reductions.
#pragma unroll
    for (int r = 0; r < 8; ++r) {
      float t = fmaxf(E0[r], E1[r]);
      t = fmaxf(t, __shfl_xor(t, 1, 32));
      t = fmaxf(t, __shfl_xor(t, 2, 32));
      t = fmaxf(t, __shfl_xor(t, 4, 32));
      t = fmaxf(t, __shfl_xor(t, 8, 32));
      const float mn = fmaxf(mrun[r], t);
      const float sc = __expf(mrun[r] - mn);
      const float p0 = __expf(E0[r] - mn);
      const float p1 = __expf(E1[r] - mn);
      float rs = p0 + p1;
      rs += __shfl_xor(rs, 1, 32);
      rs += __shfl_xor(rs, 2, 32);
      rs += __shfl_xor(rs, 4, 32);
      rs += __shfl_xor(rs, 8, 32);
      lrun[r] = lrun[r] * sc + rs;
      mrun[r] = mn;
#pragma unroll
      for (int ct = 0; ct < 16; ++ct) acc[ct][r] *= sc;
      const int row = r + (hi ? 8 : 0);
      Plds[wv][row][col]      = (__bf16)p0;
      Plds[wv][row][col + 16] = (__bf16)p1;
    }

    // Fence C-layout stores -> A-layout reload (same wave, private LDS tile).
    asm volatile("s_wait_dscnt 0" ::: "memory");

    const __bf16* pb = &Plds[wv][col][0] + (hi ? 8 : 0);
    v16bf Ap = ld2(pb, pb + 16);

    // outT accumulation across all 256 channels: 16 c-subtiles,
    // B = v^T (m in two 8-chunks per lane, matching A).
#pragma unroll
    for (int ct = 0; ct < 16; ++ct) {
      const int c = ct * 16 + col;
      const __bf16* vp = vb + ((size_t)b * CC + c) * NN + m0 + (hi ? 8 : 0);
      v16bf Bv = ld2(vp, vp + 16);
      acc[ct] = wmma_bf16(Ap, Bv, acc[ct]);
    }
  }

  const float g = gamma[0];
  float inv[8];
#pragma unroll
  for (int r = 0; r < 8; ++r) inv[r] = 1.0f / lrun[r];

#pragma unroll
  for (int ct = 0; ct < 16; ++ct) {
#pragma unroll
    for (int r = 0; r < 8; ++r) {
      const int c = ct * 16 + col;
      const int n = n0 + r + (hi ? 8 : 0);
      const size_t idx = ((size_t)b * CC + c) * NN + n;
      out[idx] = g * (acc[ct][r] * inv[r]) + x[idx];
    }
  }
}

// ---------------------------------------------------------------------------
extern "C" void kernel_launch(void* const* d_in, const int* in_sizes, int n_in,
                              void* d_out, int out_size, void* d_ws, size_t ws_size,
                              hipStream_t stream) {
  (void)in_sizes; (void)n_in; (void)out_size; (void)ws_size;
  const float* x     = (const float*)d_in[0];
  const float* wq    = (const float*)d_in[1];
  const float* bq    = (const float*)d_in[2];
  const float* wk    = (const float*)d_in[3];
  const float* bk    = (const float*)d_in[4];
  const float* wv    = (const float*)d_in[5];
  const float* bv    = (const float*)d_in[6];
  const float* gamma = (const float*)d_in[7];
  float* out = (float*)d_out;

  // Workspace layout (bf16 staging buffers), ~75.6 MB total.
  char* ws = (char*)d_ws;
  size_t off = 0;
  __bf16* xt  = (__bf16*)(ws + off); off += (size_t)BB * NN * CC * 2;  // x^T
  __bf16* qT  = (__bf16*)(ws + off); off += (size_t)BB * NN * CQ * 2;
  __bf16* kT  = (__bf16*)(ws + off); off += (size_t)BB * NN * CQ * 2;
  __bf16* vbb = (__bf16*)(ws + off); off += (size_t)BB * CC * NN * 2;
  __bf16* wqb = (__bf16*)(ws + off); off += (size_t)CQ * CC * 2;
  __bf16* wkb = (__bf16*)(ws + off); off += (size_t)CQ * CC * 2;
  __bf16* wvb = (__bf16*)(ws + off); off += (size_t)CC * CC * 2;

  transpose_convert_kernel<<<dim3(NN / 32, CC / 32, BB), dim3(32, 8), 0, stream>>>(x, xt);
  f32_to_bf16_kernel<<<(CQ * CC + 255) / 256, 256, 0, stream>>>(wq, wqb, CQ * CC);
  f32_to_bf16_kernel<<<(CQ * CC + 255) / 256, 256, 0, stream>>>(wk, wkb, CQ * CC);
  f32_to_bf16_kernel<<<(CC * CC + 255) / 256, 256, 0, stream>>>(wv, wvb, CC * CC);

  proj_qk_kernel<<<dim3(NN / 16, CQ / 16, BB), 32, 0, stream>>>(xt, wqb, bq, qT);
  proj_qk_kernel<<<dim3(NN / 16, CQ / 16, BB), 32, 0, stream>>>(xt, wkb, bk, kT);
  proj_v_kernel<<<dim3(NN / 16, CC / 16, BB), 32, 0, stream>>>(xt, wvb, bv, vbb);

  attn_kernel<<<dim3(NN / 64, BB), 128, 0, stream>>>(qT, kT, vbb, x, gamma, out);
}
